// FeatureSelfAttentionNet_4818953306700
// MI455X (gfx1250) — matrix operand order repaired and checked by reading, not compiled, not run on hardware
//
#include <hip/hip_runtime.h>
#include <hip/hip_bf16.h>
#include <math.h>

// MI455X (gfx1250) implementation.
// Compute-bound (~0.5 TFLOP vs ~60MB min traffic) => all GEMMs through
// v_wmma_f32_16x16x32_bf16 (wave32), shared operands staged into LDS via the
// CDNA5 async global->LDS engine (ASYNCcnt), LDS-staged softmax.

typedef unsigned short u16;
typedef __attribute__((ext_vector_type(8)))  unsigned short us8;
typedef __attribute__((ext_vector_type(16))) __bf16        v16bf;
typedef __attribute__((ext_vector_type(8)))  float         v8f;

#define BB 128
#define FC 256
#define CC 256
#define HH 8
#define DD 32
#define LL 6
#define MROWS (BB * FC) /* 32768 */

__device__ __forceinline__ u16 f2bf(float f) {
    unsigned int u = __float_as_uint(f);
    u += 0x7fffu + ((u >> 16) & 1u); // round-to-nearest-even
    return (u16)(u >> 16);
}

union Frag16 { us8 h[2]; v16bf v; };

// A fragment 16x32, contiguous k order used here (not AMD's table order): lanes 0-15 hold K=0..15;
// lanes 16-31 hold K=16..31 for row M=lane&15. Two 16B contiguous loads.
__device__ __forceinline__ v16bf load_afrag_g(const u16* base, int lane) {
    int hs = (lane >= 16) ? 16 : 0;
    Frag16 f;
    f.h[0] = *(const us8*)(base + hs);
    f.h[1] = *(const us8*)(base + 8 + hs);
    return f.v;
}
// B frag, same contiguous k order as A: column N=lane&15; lanes 0-15 hold K=0..15, lanes 16-31 K=16..31.
// colbase points at (transposed) column start + k0 -> one 32B contiguous load.
__device__ __forceinline__ v16bf load_bfrag(const u16* colbase, int lane) {
    return *(const v16bf*)(const void*)(colbase + ((lane >= 16) ? 16 : 0));
}

// --------------------------------------------- CDNA5 async global->LDS (asm)
// global_load_async_to_lds_b128: copies 16B from global to the given LDS byte offset; tracked by ASYNCcnt.
__device__ __forceinline__ void async_ld_b128(unsigned lds_byte_off, const void* g) {
    unsigned long long ga = (unsigned long long)(size_t)g;
    asm volatile("global_load_async_to_lds_b128 %0, %1, off"
                 :: "v"(lds_byte_off), "v"(ga) : "memory");
}
__device__ __forceinline__ void wait_async0() {
    asm volatile("s_wait_asynccnt 0" ::: "memory");
}

// ---------------------------------------------------------------- elementwise
__global__ __launch_bounds__(256) void embed_kernel(
    const float* __restrict__ x, const float* __restrict__ col_emb,
    const float* __restrict__ pos_emb, float* __restrict__ h) {
    int i = blockIdx.x * 256 + threadIdx.x;
    if (i >= MROWS * CC) return;
    int c = i & 255;
    int t = (i >> 8) & 255;
    h[i] = x[i] + col_emb[t * CC + c] + pos_emb[(t + 1) * CC + c];
}

// W[K,N] f32 -> Wt[N,K] bf16
__global__ __launch_bounds__(256) void transpose_kernel(
    const float* __restrict__ W, u16* __restrict__ Wt, int K, int N) {
    int i = blockIdx.x * 256 + threadIdx.x;
    if (i >= N * K) return;
    int n = i / K, k = i % K;
    Wt[i] = f2bf(W[(size_t)k * N + n]);
}

__global__ __launch_bounds__(256) void glu_kernel(
    const float* __restrict__ mlp, u16* __restrict__ g) {
    int i = blockIdx.x * 256 + threadIdx.x;
    if (i >= MROWS * CC) return;
    int m = i >> 8, c = i & 255;
    float a  = mlp[(size_t)m * 512 + c];
    float gt = mlp[(size_t)m * 512 + 256 + c];
    g[i] = f2bf(a * (1.0f / (1.0f + expf(-gt))));
}

// ------------------------------------------------------------------ layernorm
__global__ __launch_bounds__(256) void ln_kernel(
    const float* __restrict__ h, const float* __restrict__ ga,
    const float* __restrict__ be, u16* __restrict__ out) {
    const int lane = threadIdx.x & 31;
    const int wave = threadIdx.x >> 5;
    const int row  = blockIdx.x * 8 + wave;
    const float* r = h + (size_t)row * CC;
    float x[8], s = 0.f;
    #pragma unroll
    for (int j = 0; j < 8; ++j) { x[j] = r[lane * 8 + j]; s += x[j]; }
    #pragma unroll
    for (int m = 16; m >= 1; m >>= 1) s += __shfl_xor(s, m, 32);
    float mean = s * (1.0f / 256.0f), v = 0.f;
    #pragma unroll
    for (int j = 0; j < 8; ++j) { float d = x[j] - mean; v += d * d; }
    #pragma unroll
    for (int m = 16; m >= 1; m >>= 1) v += __shfl_xor(v, m, 32);
    float rstd = rsqrtf(v * (1.0f / 256.0f) + 1e-5f);
    u16* o = out + (size_t)row * CC;
    #pragma unroll
    for (int j = 0; j < 8; ++j) {
        int c = lane * 8 + j;
        o[c] = f2bf((x[j] - mean) * rstd * ga[c] + be[c]);
    }
}

// ------------------------------------------------------------------ GEMM/WMMA
// D[M,N] = A[M,K](bf16) x Wt[N,K](bf16, pre-transposed) with fused epilogue.
// 256 threads = 8 waves; wave -> 16(M) x 64(N); block -> 128(M) x 64(N).
// The shared 64x32 B-tile (4KB) is staged once per K-step into LDS with
// async global->LDS loads (one b128 per lane), removing the 8x per-wave
// redundancy; waves then read fragments with ds_load_b128.
// EPI: 0=QKV scatter  1=out-proj(+col_bias,+resid)  2=GELU->bf16
//      3=f32 store(+bias)  4=resid add(+bias)
template <int EPI>
__global__ __launch_bounds__(256) void gemm_bf16(
    const u16* __restrict__ A, const u16* __restrict__ Wt,
    int M, int N, int K,
    const float* __restrict__ bias,
    float* __restrict__ f32out, u16* __restrict__ bf16out,
    const float* __restrict__ colbias,
    u16* __restrict__ kbuf, u16* __restrict__ vbuf) {
    extern __shared__ char smem[]; // 4KB B-tile [64 cols][32 k] bf16
    u16* btile = (u16*)smem;
    const unsigned lds0 = __builtin_amdgcn_groupstaticsize();

    const int lane = threadIdx.x & 31;
    const int wave = threadIdx.x >> 5;
    const int m0   = blockIdx.y * 128 + wave * 16;
    const int n0   = blockIdx.x * 64;
    const int mrow = m0 + (lane & 15);
    const int tnl  = threadIdx.x >> 2;   // staging: col within tile
    const int tch  = threadIdx.x & 3;    // staging: 16B chunk within 64B row

    v8f acc[4] = {};
    const u16* arow = A + (size_t)mrow * K;
    for (int k0 = 0; k0 < K; k0 += 32) {
        __syncthreads(); // previous iteration's LDS reads complete
        async_ld_b128(lds0 + (unsigned)(tnl * 64 + tch * 16),
                      Wt + (size_t)(n0 + tnl) * K + k0 + tch * 8);
        wait_async0();
        __syncthreads(); // tile visible to all waves

        __builtin_prefetch(arow + k0 + 128, 0, 3); // global_prefetch_b8
        v16bf av = load_afrag_g(arow + k0, lane);
        #pragma unroll
        for (int j = 0; j < 4; ++j) {
            int nl = j * 16 + (lane & 15);
            v16bf bv = load_bfrag(btile + nl * 32, lane); // ds_load_b128 x2
            acc[j] = __builtin_amdgcn_wmma_f32_16x16x32_bf16(
                false, av, false, bv, (short)0, acc[j], false, false);
        }
    }
    const int r0 = (lane >= 16) ? 8 : 0;
    #pragma unroll
    for (int j = 0; j < 4; ++j) {
        int col = n0 + j * 16 + (lane & 15);
        #pragma unroll
        for (int i = 0; i < 8; ++i) {
            int row = m0 + r0 + i;
            float val = acc[j][i] + bias[col];
            if (EPI == 0) { // QKV: q/k -> [b,h,t,d], v -> transposed [b,h,d,t]
                int b = row >> 8, t = row & 255;
                if (col < 256) {
                    int hh = col >> 5, d = col & 31;
                    bf16out[((size_t)(b * HH + hh) * FC + t) * DD + d] = f2bf(val);
                } else if (col < 512) {
                    int n2 = col - 256, hh = n2 >> 5, d = n2 & 31;
                    kbuf[((size_t)(b * HH + hh) * FC + t) * DD + d] = f2bf(val);
                } else {
                    int n2 = col - 512, hh = n2 >> 5, d = n2 & 31;
                    vbuf[((size_t)(b * HH + hh) * DD + d) * FC + t] = f2bf(val);
                }
            } else if (EPI == 1) {
                int t = row & 255;
                f32out[(size_t)row * CC + col] += val + colbias[t * FC + col];
            } else if (EPI == 2) {
                float gx = 0.5f * val * (1.0f + erff(val * 0.70710678118654752f));
                bf16out[(size_t)row * N + col] = f2bf(gx);
            } else if (EPI == 3) {
                f32out[(size_t)row * N + col] = val;
            } else {
                f32out[(size_t)row * CC + col] += val;
            }
        }
    }
}

// ------------------------------------------------------------------ attention
// One block per (b,h). 8 waves, each owns 32 t-rows.
// K (16KB) async-staged into LDS before scores; the same LDS region is reused
// for vT (barrier-separated) before w@v. Scores (K=D=32 => one WMMA per tile)
// in LDS f32; per-lane softmax (one row per lane, wave32); exp written bf16
// into the LOW half of the lane's own float row (write offset 2s trails read
// offset 4s under lockstep -> no hazard); 1/rowsum folded into the epilogue.
__global__ __launch_bounds__(256) void attn_kernel(
    const u16* __restrict__ q, const u16* __restrict__ k,
    const u16* __restrict__ vT, const float* __restrict__ colbias,
    u16* __restrict__ attn) {
    extern __shared__ char smem[];
    float* sc = (float*)smem;                         // [8][32][256] f32 scores
    float* rs = (float*)(smem + 8 * 32 * 256 * 4);    // [8*32] 1/rowsum
    u16*   kv = (u16*)(smem + 8 * 32 * 256 * 4 + 1024); // 16KB K then vT tile
    const unsigned kvOff = __builtin_amdgcn_groupstaticsize() + 8u * 32u * 256u * 4u + 1024u;

    const int lane = threadIdx.x & 31;
    const int wave = threadIdx.x >> 5;
    const int bh = blockIdx.x, b = bh >> 3, h = bh & 7;
    const int r0 = (lane >= 16) ? 8 : 0;
    float* scW = sc + wave * 32 * 256;
    const float scale = 0.17677669529663689f; // 1/sqrt(32)

    const u16* qbh = q  + (size_t)bh * FC * DD;
    const u16* kbh = k  + (size_t)bh * FC * DD;
    const u16* vbh = vT + (size_t)bh * DD * FC;

    // Stage K[256][32] bf16 -> LDS (each thread: its 64B row via 4 async b128)
    {
        int s = threadIdx.x;
        #pragma unroll
        for (int ch = 0; ch < 4; ++ch)
            async_ld_b128(kvOff + (unsigned)(s * 64 + ch * 16), kbh + s * DD + ch * 8);
        wait_async0();
    }
    __syncthreads();

    // Phase 1: scores = scale * q k^T + col_bias
    #pragma unroll
    for (int rt = 0; rt < 2; ++rt) {
        int t0 = wave * 32 + rt * 16;
        int m  = t0 + (lane & 15);
        v16bf av = load_afrag_g(qbh + m * DD, lane);
        for (int ct = 0; ct < 16; ++ct) {
            int s0 = ct * 16;
            int n  = s0 + (lane & 15);
            v16bf bv = load_bfrag(kv + n * DD, lane); // ds_load_b128 x2
            v8f acc = {};
            acc = __builtin_amdgcn_wmma_f32_16x16x32_bf16(
                false, av, false, bv, (short)0, acc, false, false);
            #pragma unroll
            for (int i = 0; i < 8; ++i) {
                int t = t0 + r0 + i, s = s0 + (lane & 15);
                scW[(t - wave * 32) * 256 + s] = acc[i] * scale + colbias[t * FC + s];
            }
        }
    }
    __syncthreads(); // all waves done reading K tile before vT overwrites it

    // Phase 2: softmax, one row per lane (wave-local)
    {
        float* row = scW + lane * 256;
        float mx = -3.4e38f;
        for (int s = 0; s < 256; ++s) mx = fmaxf(mx, row[s]);
        float sum = 0.f;
        for (int s = 0; s < 256; ++s) sum += expf(row[s] - mx);
        rs[wave * 32 + lane] = 1.0f / sum;
        u16* wrow = (u16*)row; // bf16 overlay, low half of own f32 row
        for (int s = 0; s < 256; ++s) {
            float val = row[s];
            wrow[s] = f2bf(expf(val - mx));
        }
    }

    // Stage vT[32][256] bf16 -> same LDS region (thread: 64B via 4 async b128)
    {
        int d = threadIdx.x >> 3, part = threadIdx.x & 7;
        #pragma unroll
        for (int ch = 0; ch < 4; ++ch)
            async_ld_b128(kvOff + (unsigned)(d * 512 + part * 64 + ch * 16),
                          vbh + d * FC + part * 32 + ch * 8);
        wait_async0();
    }
    __syncthreads();

    // Phase 3: attn = (exp-weights @ v) * (1/rowsum)
    #pragma unroll
    for (int rt = 0; rt < 2; ++rt) {
        int t0 = wave * 32 + rt * 16;
        #pragma unroll
        for (int ct = 0; ct < 2; ++ct) {
            int d0 = ct * 16;
            v8f acc = {};
            for (int sb = 0; sb < 8; ++sb) {
                int s0 = sb * 32;
                int lr = rt * 16 + (lane & 15);
                const u16* wr = (const u16*)(scW + lr * 256); // bf16 overlay rows
                v16bf av = load_afrag_g(wr + s0, lane);       // ds_load_b128 x2
                int n = d0 + (lane & 15);
                v16bf bv = load_bfrag(kv + n * FC + s0, lane); // vT from LDS
                acc = __builtin_amdgcn_wmma_f32_16x16x32_bf16(
                    false, av, false, bv, (short)0, acc, false, false);
            }
            #pragma unroll
            for (int i = 0; i < 8; ++i) {
                int t = t0 + r0 + i;
                float val = acc[i] * rs[wave * 32 + (rt * 16 + r0 + i)];
                attn[((size_t)(b * FC) + t) * CC + h * DD + d0 + (lane & 15)] = f2bf(val);
            }
        }
    }
}

// ---------------------------------------------------------- final LN + mean_t
__global__ __launch_bounds__(256) void final_kernel(
    const float* __restrict__ h, const float* __restrict__ ga,
    const float* __restrict__ be, float* __restrict__ out) {
    __shared__ float red[256];
    const int lane = threadIdx.x & 31;
    const int wave = threadIdx.x >> 5;
    const int b = blockIdx.x;
    red[threadIdx.x] = 0.f;
    __syncthreads();
    float acc[8] = {0, 0, 0, 0, 0, 0, 0, 0};
    for (int t = wave; t < FC; t += 8) {
        const float* r = h + ((size_t)b * FC + t) * CC;
        float x[8], s = 0.f;
        #pragma unroll
        for (int j = 0; j < 8; ++j) { x[j] = r[lane * 8 + j]; s += x[j]; }
        #pragma unroll
        for (int m = 16; m >= 1; m >>= 1) s += __shfl_xor(s, m, 32);
        float mean = s * (1.0f / 256.0f), v = 0.f;
        #pragma unroll
        for (int j = 0; j < 8; ++j) { float d = x[j] - mean; v += d * d; }
        #pragma unroll
        for (int m = 16; m >= 1; m >>= 1) v += __shfl_xor(v, m, 32);
        float rstd = rsqrtf(v * (1.0f / 256.0f) + 1e-5f);
        #pragma unroll
        for (int j = 0; j < 8; ++j) {
            int c = lane * 8 + j;
            acc[j] += (x[j] - mean) * rstd * ga[c] + be[c];
        }
    }
    #pragma unroll
    for (int j = 0; j < 8; ++j) atomicAdd(&red[lane * 8 + j], acc[j]);
    __syncthreads();
    if (wave == 0) {
        #pragma unroll
        for (int j = 0; j < 8; ++j) {
            int c = lane * 8 + j;
            out[(size_t)b * CC + c] = red[c] * (1.0f / 256.0f);
        }
    }
}

// ---------------------------------------------------------------------- host
extern "C" void kernel_launch(void* const* d_in, const int* in_sizes, int n_in,
                              void* d_out, int out_size, void* d_ws, size_t ws_size,
                              hipStream_t stream) {
    (void)in_sizes; (void)n_in; (void)out_size; (void)ws_size;
    const float* x        = (const float*)d_in[0];
    const float* col_emb  = (const float*)d_in[1];
    const float* pos_emb  = (const float*)d_in[2];
    const float* col_bias = (const float*)d_in[3];
    const float* ln1_g = (const float*)d_in[4];
    const float* ln1_b = (const float*)d_in[5];
    const float* qkv_w = (const float*)d_in[6];
    const float* qkv_b = (const float*)d_in[7];
    const float* out_w = (const float*)d_in[8];
    const float* out_b = (const float*)d_in[9];
    const float* ln2_g = (const float*)d_in[10];
    const float* ln2_b = (const float*)d_in[11];
    const float* w1 = (const float*)d_in[12];
    const float* b1 = (const float*)d_in[13];
    const float* w2 = (const float*)d_in[14];
    const float* b2 = (const float*)d_in[15];
    const float* w3 = (const float*)d_in[16];
    const float* b3 = (const float*)d_in[17];
    const float* fin_g = (const float*)d_in[18];
    const float* fin_b = (const float*)d_in[19];
    float* out = (float*)d_out;

    char* ws = (char*)d_ws;
    size_t off = 0;
    auto carve = [&](size_t bytes) -> void* {
        void* p = ws + off;
        off += (bytes + 255) & ~(size_t)255;
        return p;
    };
    float* hbuf  = (float*)carve((size_t)MROWS * CC * 4);
    u16*   xn    = (u16*)carve((size_t)MROWS * CC * 2);
    u16*   qb    = (u16*)carve((size_t)BB * HH * FC * DD * 2);
    u16*   kb    = (u16*)carve((size_t)BB * HH * FC * DD * 2);
    u16*   vTb   = (u16*)carve((size_t)BB * HH * DD * FC * 2);
    u16*   attnb = (u16*)carve((size_t)MROWS * CC * 2);
    u16*   ub    = (u16*)carve((size_t)MROWS * 1024 * 2);
    float* mlpf  = (float*)carve((size_t)MROWS * 512 * 4);
    u16*   gbuf  = (u16*)carve((size_t)MROWS * CC * 2);
    u16*   wqkvT = (u16*)carve((size_t)LL * 768 * 256 * 2);
    u16*   woutT = (u16*)carve((size_t)LL * 256 * 256 * 2);
    u16*   w1T   = (u16*)carve((size_t)LL * 1024 * 256 * 2);
    u16*   w2T   = (u16*)carve((size_t)LL * 512 * 1024 * 2);
    u16*   w3T   = (u16*)carve((size_t)LL * 256 * 256 * 2);

    // scores 256KB + rowsums 1KB + 16KB K/vT tile, < 320KB WGP LDS
    const size_t SMEM_ATTN = (size_t)8 * 32 * 256 * 4 + 1024 + 16384;
    hipFuncSetAttribute(reinterpret_cast<const void*>(attn_kernel),
                        hipFuncAttributeMaxDynamicSharedMemorySize, (int)SMEM_ATTN);
    const size_t SMEM_GEMM = 4096;

    // embeddings into the f32 residual stream
    embed_kernel<<<(MROWS * CC + 255) / 256, 256, 0, stream>>>(x, col_emb, pos_emb, hbuf);

    // pre-transpose all weights to bf16 [N,K]
    for (int l = 0; l < LL; ++l) {
        transpose_kernel<<<(768 * 256 + 255) / 256, 256, 0, stream>>>(
            qkv_w + (size_t)l * 256 * 768, wqkvT + (size_t)l * 768 * 256, 256, 768);
        transpose_kernel<<<(256 * 256 + 255) / 256, 256, 0, stream>>>(
            out_w + (size_t)l * 256 * 256, woutT + (size_t)l * 256 * 256, 256, 256);
        transpose_kernel<<<(1024 * 256 + 255) / 256, 256, 0, stream>>>(
            w1 + (size_t)l * 256 * 1024, w1T + (size_t)l * 1024 * 256, 256, 1024);
        transpose_kernel<<<(512 * 1024 + 255) / 256, 256, 0, stream>>>(
            w2 + (size_t)l * 1024 * 512, w2T + (size_t)l * 512 * 1024, 1024, 512);
        transpose_kernel<<<(256 * 256 + 255) / 256, 256, 0, stream>>>(
            w3 + (size_t)l * 256 * 256, w3T + (size_t)l * 256 * 256, 256, 256);
    }

    for (int l = 0; l < LL; ++l) {
        ln_kernel<<<MROWS / 8, 256, 0, stream>>>(hbuf, ln1_g + l * 256, ln1_b + l * 256, xn);
        gemm_bf16<0><<<dim3(768 / 64, MROWS / 128), 256, SMEM_GEMM, stream>>>(
            xn, wqkvT + (size_t)l * 768 * 256, MROWS, 768, 256,
            qkv_b + (size_t)l * 768, nullptr, qb, nullptr, kb, vTb);
        attn_kernel<<<BB * HH, 256, SMEM_ATTN, stream>>>(qb, kb, vTb, col_bias, attnb);
        gemm_bf16<1><<<dim3(256 / 64, MROWS / 128), 256, SMEM_GEMM, stream>>>(
            attnb, woutT + (size_t)l * 256 * 256, MROWS, 256, 256,
            out_b + (size_t)l * 256, hbuf, nullptr, col_bias, nullptr, nullptr);
        ln_kernel<<<MROWS / 8, 256, 0, stream>>>(hbuf, ln2_g + l * 256, ln2_b + l * 256, xn);
        gemm_bf16<2><<<dim3(1024 / 64, MROWS / 128), 256, SMEM_GEMM, stream>>>(
            xn, w1T + (size_t)l * 1024 * 256, MROWS, 1024, 256,
            b1 + (size_t)l * 1024, nullptr, ub, nullptr, nullptr, nullptr);
        gemm_bf16<3><<<dim3(512 / 64, MROWS / 128), 256, SMEM_GEMM, stream>>>(
            ub, w2T + (size_t)l * 512 * 1024, MROWS, 512, 1024,
            b2 + (size_t)l * 512, mlpf, nullptr, nullptr, nullptr, nullptr);
        glu_kernel<<<(MROWS * CC + 255) / 256, 256, 0, stream>>>(mlpf, gbuf);
        gemm_bf16<4><<<dim3(256 / 64, MROWS / 128), 256, SMEM_GEMM, stream>>>(
            gbuf, w3T + (size_t)l * 256 * 256, MROWS, 256, 256,
            b3 + (size_t)l * 256, hbuf, nullptr, nullptr, nullptr, nullptr);
    }

    final_kernel<<<BB, 256, 0, stream>>>(hbuf, fin_g, fin_b, out);
}
